// HeteroGraphSAGE_52183852646753
// MI455X (gfx1250) — matrix operand order repaired and checked by reading, not compiled, not run on hardware
//
#include <hip/hip_runtime.h>
#include <stdint.h>

typedef __attribute__((ext_vector_type(16))) _Float16 v16h;
typedef __attribute__((ext_vector_type(8)))  _Float16 v8h;
typedef __attribute__((ext_vector_type(8)))  float    v8f;

#define NN    20000
#define DEG   16
#define F_IN  96
#define T_EMB 32
#define HDIM  128
#define G4    512           // 4*H
#define WSTR  136           // padded f16 row stride (halfs) -> 272B rows, conflict-free b128 frags
#define GSTR  516           // padded f32 row stride for gates staging

#define WIH_HALFS (G4 * WSTR)      // 69632 halfs per [512][136] matrix
#define WL_HALFS  (HDIM * WSTR)    // 17408 halfs per [128][136] matrix
#define LAYER_W_HALFS (2 * WIH_HALFS + 2 * WL_HALFS)

#define LSTM_SMEM (2*WIH_HALFS*2 + 3*(16*WSTR*2) + G4*4 + 16*GSTR*4)   // 326,656 B
#define LIN_SMEM  (2*WL_HALFS*2 + 2*(16*WSTR*2))                       //  78,336 B

__device__ __forceinline__ float sig_fast(float x){
  return __builtin_amdgcn_rcpf(1.0f + __expf(-x));
}
__device__ __forceinline__ float tanh_fast(float x){
#if __has_builtin(__builtin_amdgcn_tanhf)
  return __builtin_amdgcn_tanhf(x);           // gfx1250 V_TANH_F32
#else
  float e = __expf(-2.0f * x);
  return (1.0f - e) * __builtin_amdgcn_rcpf(1.0f + e);
#endif
}

__device__ __forceinline__ uint32_t lds_off(const void* p){
  // generic LDS pointers carry the workgroup-relative offset in the low 32 bits
  return (uint32_t)(uintptr_t)p;
}
// CDNA5 async global->LDS copy (16B per lane), tracked with ASYNCcnt
__device__ __forceinline__ void async_cp16(uint32_t lds_dst, const void* gsrc){
  asm volatile("global_load_async_to_lds_b128 %0, %1, off"
               :: "v"(lds_dst), "v"((uint64_t)(uintptr_t)gsrc) : "memory");
}
__device__ __forceinline__ void wait_async0(){
  asm volatile("s_wait_asynccnt 0x0" ::: "memory");
}

// A-matrix 16x32 f16 fragment: half-wave 0 holds K 0..15, half-wave 1 holds K 16..31
__device__ __forceinline__ v16h fragA_ld(const _Float16* rowbase, int kbase, int lane){
  const int off = (lane & 16) ? 16 : 0;
  v8h lo = *(const v8h*)(rowbase + kbase + off);
  v8h hi = *(const v8h*)(rowbase + kbase + off + 8);
  v16h r;
#pragma unroll
  for (int i = 0; i < 8; ++i){ r[i] = lo[i]; r[i+8] = hi[i]; }
  return r;
}
// B-matrix 32x16 f16 fragment: half-wave 0 holds K 0..15, half-wave 1 holds K 16..31
__device__ __forceinline__ v16h fragB_ld(const _Float16* rowbase, int kbase, int lane){
  const int off = (lane & 16) ? 16 : 0;
  v8h lo = *(const v8h*)(rowbase + kbase + off);
  v8h hi = *(const v8h*)(rowbase + kbase + off + 8);
  v16h r;
#pragma unroll
  for (int i = 0; i < 8; ++i){ r[i] = lo[i]; r[i+8] = hi[i]; }
  return r;
}

__global__ void hgs_embed(const float* __restrict__ x, const int* __restrict__ tids,
                          const float* __restrict__ emb, _Float16* __restrict__ out){
  int i = blockIdx.x * blockDim.x + threadIdx.x;
  if (i >= NN * HDIM) return;
  int n = i >> 7, k = i & 127;
  float v = (k < F_IN) ? x[n * F_IN + k] : emb[tids[n] * T_EMB + (k - F_IN)];
  out[i] = (_Float16)v;
}

// f32 [rows][128] -> f16 [rows][136] (padded, zero pad cols) : once per weight matrix
__global__ void hgs_cvtpad(const float* __restrict__ src, _Float16* __restrict__ dst, int rows){
  int i = blockIdx.x * blockDim.x + threadIdx.x;
  if (i >= rows * WSTR) return;
  int r = i / WSTR, c = i - r * WSTR;
  dst[i] = (c < HDIM) ? (_Float16)src[r * HDIM + c] : (_Float16)0.0f;
}

__global__ __launch_bounds__(256, 1)
void hgs_lstm(const _Float16* __restrict__ hin, const int* __restrict__ src,
              const _Float16* __restrict__ w16,   // wih||whh, pre-padded f16, 2*WIH_HALFS
              const float* __restrict__ b_ih, const float* __restrict__ b_hh,
              _Float16* __restrict__ hagg){
  extern __shared__ char smem_raw[];
  _Float16* s_wih = (_Float16*)smem_raw;            // [512][136]
  _Float16* s_whh = s_wih + WIH_HALFS;              // [512][136]
  _Float16* s_xt0 = s_whh + WIH_HALFS;              // [16][136]
  _Float16* s_xt1 = s_xt0 + 16 * WSTR;              // [16][136]
  _Float16* s_h   = s_xt1 + 16 * WSTR;              // [16][136]
  float*    s_bias = (float*)(s_h + 16 * WSTR);     // [512]
  float*    s_g    = s_bias + G4;                   // [16][516]

  const int tid  = threadIdx.x;
  const int lane = tid & 31;
  const int wv   = tid >> 5;
  const int base = blockIdx.x * 16;
  const int r    = tid >> 4;          // node-in-tile this thread serves for gather/update
  const int part = tid & 15;          // 8-half chunk within a 128-half row
  const int rowbase = (base + r) * DEG;

  // ---- async stage weights: linear global->LDS copy of wih||whh (278,528 B) ----
  {
    const uint32_t l0 = lds_off(s_wih);
    const char* g0 = (const char*)w16;
    for (int ofs = tid * 16; ofs < 2 * WIH_HALFS * 2; ofs += 256 * 16)
      async_cp16(l0 + ofs, g0 + ofs);
  }
  // ---- software-pipelined gather: issue t=0 now, keep idx for t=1 in a register ----
  const uint32_t xt0_a = lds_off(s_xt0) + (uint32_t)(r * WSTR + part * 8) * 2;
  const uint32_t xt1_a = lds_off(s_xt1) + (uint32_t)(r * WSTR + part * 8) * 2;
  int idx0     = src[rowbase + 0];
  int idx_next = src[rowbase + 1];
  async_cp16(xt0_a, hin + (size_t)idx0 * HDIM + part * 8);

  for (int j = tid; j < G4; j += 256) s_bias[j] = b_ih[j] + b_hh[j];
  for (int j = tid; j < 16 * WSTR; j += 256) s_h[j] = (_Float16)0.0f;

  const int nd = tid >> 4;
  const int j8 = (tid & 15) * 8;
  float c[8];
#pragma unroll
  for (int u = 0; u < 8; ++u) c[u] = 0.0f;

  const int m = lane & 15;
  const _Float16* arow_x0 = s_xt0 + m * WSTR;
  const _Float16* arow_x1 = s_xt1 + m * WSTR;
  const _Float16* arow_h  = s_h   + m * WSTR;

  // weights + xt0 resident; all LDS init visible
  wait_async0();
  __syncthreads();

  // ---- hoist the 16 time-invariant W_hh B-fragments into registers (128 VGPRs) ----
  v16h bhh[16];
#pragma unroll
  for (int q = 0; q < 4; ++q){
    const _Float16* brow = s_whh + ((wv + q * 8) * 16 + m) * WSTR;
#pragma unroll
    for (int kc = 0; kc < 4; ++kc)
      bhh[q * 4 + kc] = fragB_ld(brow, kc * 32, lane);
  }

#pragma unroll 2
  for (int t = 0; t < DEG; ++t){
    // issue gather for t+1 into the other buffer; prefetch index for t+2
    if (t + 1 < DEG){
      uint32_t dst = ((t + 1) & 1) ? xt1_a : xt0_a;
      async_cp16(dst, hin + (size_t)idx_next * HDIM + part * 8);
      idx_next = (t + 2 < DEG) ? src[rowbase + t + 2] : 0;
    }
    const _Float16* arow_x = (t & 1) ? arow_x1 : arow_x0;

    // gates[16][512] = x_t @ W_ih^T + h @ W_hh^T
    // 4 concurrent accumulators; A-frags hoisted per k-chunk; W_hh frags from registers
    v8f acc[4] = {};
#pragma unroll
    for (int kc = 0; kc < 4; ++kc){
      v16h ax = fragA_ld(arow_x, kc * 32, lane);
      v16h ah = fragA_ld(arow_h, kc * 32, lane);
#pragma unroll
      for (int q = 0; q < 4; ++q){
        const int nt = wv + q * 8;
        v16h bi = fragB_ld(s_wih + (nt * 16 + m) * WSTR, kc * 32, lane);
        acc[q] = __builtin_amdgcn_wmma_f32_16x16x32_f16(false, ax, false, bi, (short)0, acc[q], false, false);
        acc[q] = __builtin_amdgcn_wmma_f32_16x16x32_f16(false, ah, false, bhh[q * 4 + kc], (short)0, acc[q], false, false);
      }
    }
    const int rb = (lane & 16) ? 8 : 0;
#pragma unroll
    for (int q = 0; q < 4; ++q){
      const int col = (wv + q * 8) * 16 + m;
#pragma unroll
      for (int rr = 0; rr < 8; ++rr) s_g[(rb + rr) * GSTR + col] = acc[q][rr];
    }
    __syncthreads();        // gates ready

    // i,f,g,o nonlinearity + cell/state update; c in registers
#pragma unroll
    for (int u = 0; u < 8; ++u){
      int j = j8 + u;
      float ig = s_g[nd * GSTR + j]            + s_bias[j];
      float fg = s_g[nd * GSTR + j + HDIM]     + s_bias[j + HDIM];
      float gg = s_g[nd * GSTR + j + 2*HDIM]   + s_bias[j + 2*HDIM];
      float og = s_g[nd * GSTR + j + 3*HDIM]   + s_bias[j + 3*HDIM];
      float cc = sig_fast(fg) * c[u] + sig_fast(ig) * tanh_fast(gg);
      c[u] = cc;
      s_h[nd * WSTR + j] = (_Float16)(sig_fast(og) * tanh_fast(cc));
    }

    // end of step: wait for next x_t gather and make s_h/s_g updates visible
    if (t + 1 < DEG){
      wait_async0();
      __syncthreads();
    }
  }

  // each thread wrote these s_h elements itself — no barrier needed
#pragma unroll
  for (int u = 0; u < 8; ++u)
    hagg[(base + nd) * HDIM + j8 + u] = s_h[nd * WSTR + j8 + u];
}

__global__ __launch_bounds__(256, 2)
void hgs_lin(const _Float16* __restrict__ hagg, const _Float16* __restrict__ xin,
             const _Float16* __restrict__ wlr16,   // wl||wr, pre-padded f16, 2*WL_HALFS
             const float* __restrict__ b_l,
             _Float16* __restrict__ out16, float* __restrict__ out32, int relu){
  extern __shared__ char smem_raw[];
  _Float16* s_wl = (_Float16*)smem_raw;     // [128][136]
  _Float16* s_wr = s_wl + WL_HALFS;         // [128][136]
  _Float16* s_a  = s_wr + WL_HALFS;         // [16][136]
  _Float16* s_x  = s_a + 16 * WSTR;         // [16][136]

  const int tid = threadIdx.x, lane = tid & 31, wv = tid >> 5;
  const int base = blockIdx.x * 16;
  const int r = tid >> 4, part = tid & 15;

  // async stage weights (linear) + both A tiles
  {
    const uint32_t l0 = lds_off(s_wl);
    const char* g0 = (const char*)wlr16;
    for (int ofs = tid * 16; ofs < 2 * WL_HALFS * 2; ofs += 256 * 16)
      async_cp16(l0 + ofs, g0 + ofs);
    const uint32_t ta = (uint32_t)(r * WSTR + part * 8) * 2;
    async_cp16(lds_off(s_a) + ta, hagg + (size_t)(base + r) * HDIM + part * 8);
    async_cp16(lds_off(s_x) + ta, xin  + (size_t)(base + r) * HDIM + part * 8);
  }
  wait_async0();
  __syncthreads();

  const int m = lane & 15;
  v8f acc_l = {}, acc_r = {};
  const _Float16* brow_l = s_wl + (wv * 16 + m) * WSTR;
  const _Float16* brow_r = s_wr + (wv * 16 + m) * WSTR;
  const _Float16* arow_a = s_a + m * WSTR;
  const _Float16* arow_x = s_x + m * WSTR;
#pragma unroll
  for (int kc = 0; kc < 4; ++kc){
    v16h a  = fragA_ld(arow_a, kc * 32, lane);
    v16h b  = fragB_ld(brow_l, kc * 32, lane);
    acc_l = __builtin_amdgcn_wmma_f32_16x16x32_f16(false, a, false, b, (short)0, acc_l, false, false);
    v16h a2 = fragA_ld(arow_x, kc * 32, lane);
    v16h b2 = fragB_ld(brow_r, kc * 32, lane);
    acc_r = __builtin_amdgcn_wmma_f32_16x16x32_f16(false, a2, false, b2, (short)0, acc_r, false, false);
  }
  const int col   = wv * 16 + m;
  const int rbase = (lane & 16) ? 8 : 0;
  const float bias = b_l[col];
#pragma unroll
  for (int rr = 0; rr < 8; ++rr){
    float v = acc_l[rr] + acc_r[rr] + bias;
    if (relu) v = fmaxf(v, 0.0f);
    int node = base + rbase + rr;
    if (out32) out32[node * HDIM + col] = v;
    else       out16[node * HDIM + col] = (_Float16)v;
  }
}

extern "C" void kernel_launch(void* const* d_in, const int* in_sizes, int n_in,
                              void* d_out, int out_size, void* d_ws, size_t ws_size,
                              hipStream_t stream){
  (void)in_sizes; (void)n_in; (void)out_size; (void)ws_size;
  const float* x     = (const float*)d_in[0];
  const int*   tids  = (const int*)d_in[1];
  const int*   src   = (const int*)d_in[2];
  const float* emb   = (const float*)d_in[4];

  const float* W[3][7];
  for (int l = 0; l < 3; ++l)
    for (int p = 0; p < 7; ++p)
      W[l][p] = (const float*)d_in[5 + l * 7 + p];   // w_ih, w_hh, b_ih, b_hh, w_l, b_l, w_r

  _Float16* bufA = (_Float16*)d_ws;
  _Float16* bufB = bufA + (size_t)NN * HDIM;
  _Float16* bufC = bufB + (size_t)NN * HDIM;
  _Float16* wbase = bufC + (size_t)NN * HDIM;        // pre-converted padded f16 weights
  float* out = (float*)d_out;

  hipFuncSetAttribute((const void*)hgs_lstm, hipFuncAttributeMaxDynamicSharedMemorySize, LSTM_SMEM);
  hipFuncSetAttribute((const void*)hgs_lin,  hipFuncAttributeMaxDynamicSharedMemorySize, LIN_SMEM);

  const int tiles = NN / 16;   // 1250

  // pre-convert all weights to padded f16 (once)
  _Float16* wih16[3]; _Float16* wlr16[3];
  for (int l = 0; l < 3; ++l){
    _Float16* p = wbase + (size_t)l * LAYER_W_HALFS;
    wih16[l] = p;                                    // wih || whh
    wlr16[l] = p + 2 * WIH_HALFS;                    // wl  || wr
    hgs_cvtpad<<<(G4 * WSTR + 255) / 256, 256, 0, stream>>>(W[l][0], p, G4);
    hgs_cvtpad<<<(G4 * WSTR + 255) / 256, 256, 0, stream>>>(W[l][1], p + WIH_HALFS, G4);
    hgs_cvtpad<<<(HDIM * WSTR + 255) / 256, 256, 0, stream>>>(W[l][4], wlr16[l], HDIM);
    hgs_cvtpad<<<(HDIM * WSTR + 255) / 256, 256, 0, stream>>>(W[l][6], wlr16[l] + WL_HALFS, HDIM);
  }

  hgs_embed<<<(NN * HDIM + 255) / 256, 256, 0, stream>>>(x, tids, emb, bufA);

  // Layer 1: in=bufA -> agg=bufB -> out=bufC (f16, relu)
  hgs_lstm<<<tiles, 256, LSTM_SMEM, stream>>>(bufA, src, wih16[0], W[0][2], W[0][3], bufB);
  hgs_lin <<<tiles, 256, LIN_SMEM,  stream>>>(bufB, bufA, wlr16[0], W[0][5], bufC, nullptr, 1);

  // Layer 2: in=bufC -> agg=bufB -> out=bufA (f16, relu)
  hgs_lstm<<<tiles, 256, LSTM_SMEM, stream>>>(bufC, src, wih16[1], W[1][2], W[1][3], bufB);
  hgs_lin <<<tiles, 256, LIN_SMEM,  stream>>>(bufB, bufC, wlr16[1], W[1][5], bufA, nullptr, 1);

  // Layer 3: in=bufA -> agg=bufB -> out=d_out (f32, no relu)
  hgs_lstm<<<tiles, 256, LSTM_SMEM, stream>>>(bufA, src, wih16[2], W[2][2], W[2][3], bufB);
  hgs_lin <<<tiles, 256, LIN_SMEM,  stream>>>(bufB, bufA, wlr16[2], W[2][5], nullptr, out, 0);
}
